// Vit_Spatial_Map_1992864825604
// MI455X (gfx1250) — matrix operand order repaired and checked by reading, not compiled, not run on hardware
//
#include <hip/hip_runtime.h>

typedef __attribute__((ext_vector_type(16))) __bf16 v16bf;
typedef __attribute__((ext_vector_type(8)))  __bf16 v8bf;
typedef __attribute__((ext_vector_type(8)))  float  v8f;

#define DIMc 256
#define PDc  4096

static __device__ __forceinline__ v16bf cat8(v8bf lo, v8bf hi) {
  return __builtin_shufflevector(lo, hi, 0,1,2,3,4,5,6,7,8,9,10,11,12,13,14,15);
}

static __device__ __forceinline__ v8f wmma_bf16(v16bf a, v16bf b, v8f c) {
  return __builtin_amdgcn_wmma_f32_16x16x32_bf16(false, a, false, b, (short)0, c, false, false);
}

// B fragment of W^T (K x N tile) from row-major W (Ncols x K) bf16.
// Lane l holds column n0+(l%16), elements 0..7 = K k0+off..+7, 8..15 = K k0+off+16..+23, off=(l<16?0:8).
static __device__ __forceinline__ v16bf load_bfrag(const __bf16* __restrict__ W,
                                                   int K, int n0, int k0, int lane) {
  int n  = n0 + (lane & 15);
  int kb = k0 + ((lane & 16) ? 8 : 0);
  const __bf16* p = W + (size_t)n * K + kb;
  v8bf lo = *(const v8bf*)p;
  v8bf hi = *(const v8bf*)(p + 16);
  return cat8(lo, hi);
}

// A fragment from row-major (M x K) bf16: lane l row m0+(l%16),
// elements 0..7 = K off..off+7, 8..15 = K off+16..off+23, off=(l<16?0:8).
static __device__ __forceinline__ v16bf load_afrag(const __bf16* __restrict__ A,
                                                   int K, int m0, int k0, int lane) {
  int m   = m0 + (lane & 15);
  int off = (lane & 16) ? 8 : 0;
  const __bf16* p = A + (size_t)m * K + k0 + off;
  v8bf g0 = *(const v8bf*)p;
  v8bf g1 = *(const v8bf*)(p + 16);
  return cat8(g0, g1);
}

// A fragment for the embed GEMM, loaded straight from x with the unfold
// (patch extraction) fused in: K index = c*64 + kh*8 + kw, so each 8-element
// K group is one contiguous 8-float (32 B) run of an image row. f32->bf16 in regs.
static __device__ __forceinline__ v16bf load_afrag_x(const float* __restrict__ x,
                                                     int m0, int k0, int lane) {
  int grow = m0 + (lane & 15);
  int bn = grow >> 10, np = grow & 1023;
  int ph = np >> 5, pw = np & 31;
  int k  = k0 + ((lane & 16) ? 8 : 0);         // multiple of 8
  int c  = k >> 6, kh = (k & 63) >> 3;         // kh in {0,1,4,5}; +2 row for K+16
  const float* p = x + (size_t)((((bn << 6) + c) << 8) + (ph << 3) + kh) * 256 + (pw << 3);
  float4 f0 = *(const float4*)(p);
  float4 f1 = *(const float4*)(p + 4);
  float4 f2 = *(const float4*)(p + 512);       // kh+2 image row
  float4 f3 = *(const float4*)(p + 516);
  v16bf a;
  a[0]=(__bf16)f0.x;  a[1]=(__bf16)f0.y;  a[2]=(__bf16)f0.z;  a[3]=(__bf16)f0.w;
  a[4]=(__bf16)f1.x;  a[5]=(__bf16)f1.y;  a[6]=(__bf16)f1.z;  a[7]=(__bf16)f1.w;
  a[8]=(__bf16)f2.x;  a[9]=(__bf16)f2.y;  a[10]=(__bf16)f2.z; a[11]=(__bf16)f2.w;
  a[12]=(__bf16)f3.x; a[13]=(__bf16)f3.y; a[14]=(__bf16)f3.z; a[15]=(__bf16)f3.w;
  return a;
}

// Bandwidth-only weight conversion: 8 f32 -> 8 bf16 per thread (b128 in, b128 out).
__global__ void k_cvt8(const float* __restrict__ s, __bf16* __restrict__ d, int n) {
  int i = (blockIdx.x * 256 + threadIdx.x) << 3;
  if (i < n) {
    float4 f0 = *(const float4*)(s + i);
    float4 f1 = *(const float4*)(s + i + 4);
    v8bf r;
    r[0]=(__bf16)f0.x; r[1]=(__bf16)f0.y; r[2]=(__bf16)f0.z; r[3]=(__bf16)f0.w;
    r[4]=(__bf16)f1.x; r[5]=(__bf16)f1.y; r[6]=(__bf16)f1.z; r[7]=(__bf16)f1.w;
    *(v8bf*)(d + i) = r;
  }
}

// GEMM1: emb = unfold(x) @ W_embed^T + b + pos -> bf16. Wave tile 16x64,
// 4 waves cover N=256 per m-tile (shared A via L1), 2 m-tiles per block.
// A fragment (long-latency x path) double-buffered; the 4 current-iteration
// B fragments live in distinct registers so all 8 B loads issue as one clause
// and the 4 WMMAs drain them with partial loadcnt waits.
__global__ __launch_bounds__(256, 1)
void k_embed(const float* __restrict__ x, const __bf16* __restrict__ Wemb,
             const float* __restrict__ bemb, const float* __restrict__ pos,
             __bf16* __restrict__ emb) {
  int tid = threadIdx.x, lane = tid & 31, w = tid >> 5;
  int m0 = (blockIdx.x * 2 + (w >> 2)) << 4;
  int n0 = (w & 3) << 6;
  v8f acc[4] = {};
  v16bf a = load_afrag_x(x, m0, 0, lane);
  for (int k0 = 0; k0 < PDc - 32; k0 += 32) {
    v16bf b0 = load_bfrag(Wemb, PDc, n0,      k0, lane);
    v16bf b1 = load_bfrag(Wemb, PDc, n0 + 16, k0, lane);
    v16bf b2 = load_bfrag(Wemb, PDc, n0 + 32, k0, lane);
    v16bf b3 = load_bfrag(Wemb, PDc, n0 + 48, k0, lane);
    v16bf an = load_afrag_x(x, m0, k0 + 32, lane);   // prefetch next A
    acc[0] = wmma_bf16(a, b0, acc[0]);
    acc[1] = wmma_bf16(a, b1, acc[1]);
    acc[2] = wmma_bf16(a, b2, acc[2]);
    acc[3] = wmma_bf16(a, b3, acc[3]);
    a = an;
  }
  {
    v16bf b0 = load_bfrag(Wemb, PDc, n0,      PDc - 32, lane);
    v16bf b1 = load_bfrag(Wemb, PDc, n0 + 16, PDc - 32, lane);
    v16bf b2 = load_bfrag(Wemb, PDc, n0 + 32, PDc - 32, lane);
    v16bf b3 = load_bfrag(Wemb, PDc, n0 + 48, PDc - 32, lane);
    acc[0] = wmma_bf16(a, b0, acc[0]);
    acc[1] = wmma_bf16(a, b1, acc[1]);
    acc[2] = wmma_bf16(a, b2, acc[2]);
    acc[3] = wmma_bf16(a, b3, acc[3]);
  }
  #pragma unroll
  for (int f = 0; f < 4; ++f) {
    int col = n0 + f * 16 + (lane & 15);
    float bias = bemb[col];
    #pragma unroll
    for (int i = 0; i < 8; ++i) {
      int grow = m0 + i + ((lane & 16) ? 8 : 0);
      int np = grow & 1023;
      float v = acc[f][i] + bias + pos[np * DIMc + col];
      emb[(size_t)grow * DIMc + col] = (__bf16)v;
    }
  }
}

// Fully-unrolled K=256 MMA body (8 steps); distinct B registers per step so
// fragment loads batch into clauses instead of serializing against each WMMA.
#define MMA_K256(Asrc, Wsrc)                                                  \
  v8f acc[4] = {};                                                            \
  _Pragma("unroll")                                                           \
  for (int k0 = 0; k0 < DIMc; k0 += 32) {                                     \
    v16bf a  = load_afrag((Asrc), DIMc, m0, k0, lane);                        \
    v16bf b0 = load_bfrag((Wsrc), DIMc, n0,      k0, lane);                   \
    v16bf b1 = load_bfrag((Wsrc), DIMc, n0 + 16, k0, lane);                   \
    v16bf b2 = load_bfrag((Wsrc), DIMc, n0 + 32, k0, lane);                   \
    v16bf b3 = load_bfrag((Wsrc), DIMc, n0 + 48, k0, lane);                   \
    acc[0] = wmma_bf16(a, b0, acc[0]);                                        \
    acc[1] = wmma_bf16(a, b1, acc[1]);                                        \
    acc[2] = wmma_bf16(a, b2, acc[2]);                                        \
    acc[3] = wmma_bf16(a, b3, acc[3]);                                        \
  }

// GEMM2: q = mean over head-dim of emb @ Wq^T.  One wave = 16 rows x one head
// (64 cols = 4 frags); cross-lane xor-reduce over the 16-lane halves.
__global__ void k_q(const __bf16* __restrict__ emb, const __bf16* __restrict__ Wq,
                    float* __restrict__ qv) {
  int tid = threadIdx.x, lane = tid & 31, w = tid >> 5;
  int m0 = (blockIdx.x * 8 + w) << 4;
  int h = blockIdx.y, n0 = h << 6;
  MMA_K256(emb, Wq)
  v8f s = acc[0] + acc[1] + acc[2] + acc[3];
  #pragma unroll
  for (int i = 0; i < 8; ++i) {
    float t = s[i];
    t += __shfl_xor(t, 1, 32);
    t += __shfl_xor(t, 2, 32);
    t += __shfl_xor(t, 4, 32);
    t += __shfl_xor(t, 8, 32);
    if (lane == 0 || lane == 16) {
      int grow = m0 + i + ((lane >> 1) & 8);
      int bn = grow >> 10, np = grow & 1023;
      qv[(((bn << 2) + h) << 10) + np] = t * (1.0f / 64.0f);
    }
  }
}

// k = sigmoid(q @ Wk^T): tiny (67 MFLOP), plain f32. One wave per output row m,
// lanes = the 32 (bn,h) vectors; q tile staged transposed in LDS (bank-conflict free).
__global__ void k_ksig(const float* __restrict__ q, const float* __restrict__ Wk,
                       float* __restrict__ kk) {
  __shared__ float qs[4096];                       // 128 n x 32 bh
  int tid = threadIdx.x, bh = tid & 31;
  int m = blockIdx.x * 8 + (tid >> 5);
  float acc = 0.0f;
  for (int n0 = 0; n0 < 1024; n0 += 128) {
    __syncthreads();
    #pragma unroll
    for (int e = tid; e < 4096; e += 256)
      qs[e] = q[((e & 31) << 10) + n0 + (e >> 5)];
    __syncthreads();
    const float* wrow = Wk + (size_t)m * 1024 + n0;
    #pragma unroll 8
    for (int nl = 0; nl < 128; ++nl)
      acc = fmaf(qs[(nl << 5) + bh], wrow[nl], acc);
  }
  kk[(bh << 10) + m] = 1.0f / (1.0f + __expf(-acc));
}

// GEMM3: v = emb @ Wv^T, fused with the diag(k) scaling -> kv (bf16).
__global__ void k_vkv(const __bf16* __restrict__ emb, const __bf16* __restrict__ Wv,
                      const float* __restrict__ kk, __bf16* __restrict__ kv) {
  int tid = threadIdx.x, lane = tid & 31, w = tid >> 5;
  int m0 = (blockIdx.x * 8 + w) << 4;
  int h = blockIdx.y, n0 = h << 6;
  MMA_K256(emb, Wv)
  int bn = m0 >> 10, np0 = m0 & 1023;
  float kl = kk[(((bn << 2) + h) << 10) + np0 + (lane & 15)]; // lane l holds k(row l%16)
  #pragma unroll
  for (int i = 0; i < 8; ++i) {
    float kr = __shfl(kl, ((lane & 16) ? 8 : 0) + i, 32);     // broadcast row's k
    #pragma unroll
    for (int f = 0; f < 4; ++f) acc[f][i] *= kr;
  }
  #pragma unroll
  for (int f = 0; f < 4; ++f) {
    int col = n0 + f * 16 + (lane & 15);
    #pragma unroll
    for (int i = 0; i < 8; ++i) {
      int grow = m0 + i + ((lane & 16) ? 8 : 0);
      kv[(size_t)grow * DIMc + col] = (__bf16)acc[f][i];
    }
  }
}

// GEMM4: out = kv @ Wout^T + b_out, with the fold back to (B,C,H,W) fused
// into the epilogue stores (each 64-col tile = one channel's 8x8 patch).
__global__ void k_out(const __bf16* __restrict__ kv, const __bf16* __restrict__ Wout,
                      const float* __restrict__ bout, float* __restrict__ out) {
  int tid = threadIdx.x, lane = tid & 31, w = tid >> 5;
  int m0 = blockIdx.x << 4;
  int n0 = (blockIdx.y << 9) + (w << 6);
  MMA_K256(kv, Wout)
  #pragma unroll
  for (int f = 0; f < 4; ++f) {
    int col = n0 + f * 16 + (lane & 15);
    float bias = bout[col];
    int c = col >> 6, inner = col & 63, kh = inner >> 3, kw = inner & 7;
    #pragma unroll
    for (int i = 0; i < 8; ++i) {
      int grow = m0 + i + ((lane & 16) ? 8 : 0);
      int bn = grow >> 10, np = grow & 1023;
      int ph = np >> 5, pw = np & 31;
      size_t addr = (size_t)((((bn << 6) + c) << 8) + (ph << 3) + kh) * 256 + (pw << 3) + kw;
      out[addr] = acc[f][i] + bias;
    }
  }
}

extern "C" void kernel_launch(void* const* d_in, const int* in_sizes, int n_in,
                              void* d_out, int out_size, void* d_ws, size_t ws_size,
                              hipStream_t stream) {
  (void)in_sizes; (void)n_in; (void)out_size; (void)ws_size;
  const float* x    = (const float*)d_in[0];
  const float* Wemb = (const float*)d_in[1];
  const float* bemb = (const float*)d_in[2];
  const float* pos  = (const float*)d_in[3];
  const float* Wq   = (const float*)d_in[4];
  const float* Wk   = (const float*)d_in[5];
  const float* Wv   = (const float*)d_in[6];
  const float* Wout = (const float*)d_in[7];
  const float* bout = (const float*)d_in[8];
  float* out = (float*)d_out;

  char* ws = (char*)d_ws;                           // ~12.5 MB used
  __bf16* Wemb_bf = (__bf16*)(ws + 0);              // 2 MB
  __bf16* Wq_bf   = (__bf16*)(ws + 2097152);        // 128 KB
  __bf16* Wv_bf   = (__bf16*)(ws + 2228224);        // 128 KB
  __bf16* Wout_bf = (__bf16*)(ws + 2359296);        // 2 MB
  __bf16* emb_bf  = (__bf16*)(ws + 4456448);        // 4 MB
  float*  qv      = (float*) (ws + 8650752);        // 128 KB
  float*  kkv     = (float*) (ws + 8781824);        // 128 KB
  __bf16* kv_bf   = (__bf16*)(ws + 8912896);        // 4 MB

  k_cvt8<<<512, 256, 0, stream>>>(Wemb, Wemb_bf, 1048576);
  k_cvt8<<<32,  256, 0, stream>>>(Wq,   Wq_bf,   65536);
  k_cvt8<<<32,  256, 0, stream>>>(Wv,   Wv_bf,   65536);
  k_cvt8<<<512, 256, 0, stream>>>(Wout, Wout_bf, 1048576);

  k_embed<<<dim3(256),     256, 0, stream>>>(x, Wemb_bf, bemb, pos, emb_bf);
  k_q    <<<dim3(64, 4),   256, 0, stream>>>(emb_bf, Wq_bf, qv);
  k_ksig <<<dim3(128),     256, 0, stream>>>(qv, Wk, kkv);
  k_vkv  <<<dim3(64, 4),   256, 0, stream>>>(emb_bf, Wv_bf, kkv, kv_bf);
  k_out  <<<dim3(512, 8),  256, 0, stream>>>(kv_bf, Wout_bf, bout, out);
}
